// TransformerBlock_19061064860315
// MI455X (gfx1250) — matrix operand order repaired and checked by reading, not compiled, not run on hardware
//
#include <hip/hip_runtime.h>
#include <hip/hip_bf16.h>

typedef __attribute__((ext_vector_type(16))) _Float16 v16h;
typedef __attribute__((ext_vector_type(4)))  _Float16 v4h;
typedef __attribute__((ext_vector_type(8)))  float    v8f;
typedef unsigned int u32x4 __attribute__((ext_vector_type(4)));
typedef int          i32x8 __attribute__((ext_vector_type(8)));
typedef int          i32x4 __attribute__((ext_vector_type(4)));

#define S_LEN 2048
#define CH    64
#define NH    8
#define HD    64
#define TD    512
#define FFD   1024

static __device__ __forceinline__ int a_col(int i, int half) {
  // 16-bit A-matrix 16x32 layout: K = 16*(i/8) + 8*half + (i%8)
  return ((i >> 3) << 4) + (half << 3) + (i & 7);
}
static __device__ __forceinline__ int b_row(int i, int half) {
  // 16-bit B-matrix 32x16 layout: K row = 16*(i/8) + 8*half + (i%8), N col = lane%16
  return ((i >> 3) << 4) + (half << 3) + (i & 7);
}
static __device__ __forceinline__ v8f wmma_f16(v16h a, v16h b, v8f c) {
  return __builtin_amdgcn_wmma_f32_16x16x32_f16(false, a, false, b, (short)0, c,
                                                false, false);
}

// ---------------------------------------------------------------------------
// TDM: async 2-D f16 tile load Global -> LDS (D# built per ISA 8.3/8.4).
// ---------------------------------------------------------------------------
static __device__ __forceinline__ void tdm_load_2d_f16(
    unsigned lds_off, const _Float16* gaddr, unsigned tile_d0, unsigned tile_d1,
    unsigned tensor_d0, unsigned tensor_d1, unsigned stride0) {
  unsigned long long ga = (unsigned long long)(uintptr_t)gaddr;
  u32x4 g0;
  g0[0] = 1u;                                    // count=1 valid user D#
  g0[1] = lds_off;                               // lds_addr (bytes)
  g0[2] = (unsigned)(ga & 0xffffffffu);          // global_addr[31:0]
  g0[3] = (unsigned)((ga >> 32) & 0x1ffffffu)    // global_addr[56:32]
          | (2u << 30);                          // type = 2 (image)
  i32x8 g1;
  g1[0] = (int)(1u << 16);                       // data_size=1 (2B), mask=0
  g1[1] = (int)(tensor_d0 << 16);                // tensor_dim0[15:0]
  g1[2] = (int)((tensor_d0 >> 16) | (tensor_d1 << 16));
  g1[3] = (int)((tensor_d1 >> 16) | (tile_d0 << 16));
  g1[4] = (int)(tile_d1 & 0xffffu);              // tile_dim1, tile_dim2=0
  g1[5] = (int)stride0;                          // tensor_dim0_stride[31:0]
  g1[6] = 0;
  g1[7] = 0;
  i32x4 z4 = {0, 0, 0, 0};
#if defined(__clang_major__) && __clang_major__ >= 23
  i32x8 z8 = {0, 0, 0, 0, 0, 0, 0, 0};
  __builtin_amdgcn_tensor_load_to_lds(g0, g1, z4, z4, z8, 0);
#else
  __builtin_amdgcn_tensor_load_to_lds(g0, g1, z4, z4, 0);
#endif
}

// ---------------------------------------------------------------------------
// Kernel 0: one-shot f32 -> f16 weight conversion (vectorized x4).
// ---------------------------------------------------------------------------
__global__ void __launch_bounds__(256) cvt_kernel(const float* __restrict__ src,
                                                  _Float16* __restrict__ dst,
                                                  int nelem) {
  const int i = (int)(blockIdx.x * blockDim.x + threadIdx.x) * 4;
  if (i + 3 < nelem) {
    const float4 v = *(const float4*)(src + i);
    v4h o = {(_Float16)v.x, (_Float16)v.y, (_Float16)v.z, (_Float16)v.w};
    *(v4h*)(dst + i) = o;
  }
}

// ---------------------------------------------------------------------------
// Kernel 1: fused QKV projection.  One block (8 waves) per (n, which, 16-row
// tile): the 16x64 x-tile is staged through LDS once (coalesced f32 reads,
// converted once), A fragments loaded once per wave and reused across 4
// column tiles.  Weights pre-converted to f16.  Q,K stored [n,h,s,d]; V
// stored TRANSPOSED [n,h,d,s] for contiguous attention PV fragments.
// ---------------------------------------------------------------------------
__global__ void __launch_bounds__(256) qkv_kernel(
    const float* __restrict__ x,  // [2, 64, 2048]
    const _Float16* __restrict__ Wqh, const float* __restrict__ bq,
    const _Float16* __restrict__ Wkh, const float* __restrict__ bk,
    const _Float16* __restrict__ Wvh, const float* __restrict__ bv,
    _Float16* __restrict__ Q, _Float16* __restrict__ K,
    _Float16* __restrict__ V)
{
  __shared__ _Float16 xa[16 * 64];
  const int t = threadIdx.x;
  int bid = blockIdx.x;
  const int rowT  = bid & 127;  bid >>= 7;  // S/16 = 128
  const int which = bid % 3;                // Q / K / V
  const int n     = bid / 3;                // batch

  // stage x tile [16 s x 64 c]: consecutive threads read consecutive s
  for (int e = t; e < 16 * 64; e += 256) {
    const int sr = e & 15, c = e >> 4;
    xa[sr * 64 + c] = (_Float16)x[((size_t)n * CH + c) * S_LEN + rowT * 16 + sr];
  }
  __syncthreads();

  const int lane = t & 31, wave = t >> 5;
  const int half = lane >> 4, lq = lane & 15;

  // A fragments for K-chunks c=0..31 and c=32..63, loaded once, reused 4x
  v16h a0, a1;
#pragma unroll
  for (int i = 0; i < 16; ++i) {
    const int d = a_col(i, half);
    a0[i] = xa[lq * 64 + d];
    a1[i] = xa[lq * 64 + 32 + d];
  }

  const _Float16* Wh = (which == 0) ? Wqh : ((which == 1) ? Wkh : Wvh);
  const float*    bm = (which == 0) ? bq  : ((which == 1) ? bk  : bv);
  _Float16*       Om = (which == 0) ? Q   : ((which == 1) ? K   : V);

#pragma unroll
  for (int g = 0; g < 4; ++g) {
    const int colT = wave * 4 + g;        // 8 waves x 4 = 32 col tiles
    const int ocol = colT * 16 + lq;
    v16h b0, b1;
#pragma unroll
    for (int i = 0; i < 16; ++i) {
      const int c = b_row(i, half);
      b0[i] = Wh[(size_t)ocol * CH + c];
      b1[i] = Wh[(size_t)ocol * CH + 32 + c];
    }
    v8f acc = {};
    acc = wmma_f16(a0, b0, acc);
    acc = wmma_f16(a1, b1, acc);
    const float bias = bm[ocol];
    const int h = ocol >> 6, d = ocol & 63;
    if (which == 2) {
      _Float16* vrow = Om + (((size_t)n * NH + h) * HD + d) * S_LEN;
#pragma unroll
      for (int i = 0; i < 8; ++i) {
        const int s = rowT * 16 + (half << 3) + i;
        vrow[s] = (_Float16)(acc[i] + bias);
      }
    } else {
#pragma unroll
      for (int i = 0; i < 8; ++i) {
        const int s = rowT * 16 + (half << 3) + i;
        Om[(((size_t)n * NH + h) * S_LEN + s) * HD + d] =
            (_Float16)(acc[i] + bias);
      }
    }
  }
}

// ---------------------------------------------------------------------------
// Kernel 2: flash attention.  One wave per (n, h, 16-query tile).
// K (32x64) and Vt (64x32) chunks DMA'd into per-wave LDS slabs by the
// Tensor Data Mover, double-buffered (issue chunk i+1, s_wait_tensorcnt<=2
// => chunk i landed; s_wait_dscnt 0 orders TDM refill after LDS reads).
// ---------------------------------------------------------------------------
#define AW 2  // waves per block
#define SLAB (2 * 2048 + 2 * 2048 + 512)  // K dbl-buf + V dbl-buf + P, halves
__global__ void __launch_bounds__(AW * 32) attn_kernel(
    const _Float16* __restrict__ Q, const _Float16* __restrict__ K,
    const _Float16* __restrict__ V,  // V transposed [n,h,d,s]
    _Float16* __restrict__ O)        // [2, 2048, 512]
{
  __shared__ _Float16 lds[AW * SLAB];
  const int lane = threadIdx.x & 31;
  const int wid  = threadIdx.x >> 5;
  const int wave = (int)(blockIdx.x * (blockDim.x >> 5)) + wid;
  const int half = lane >> 4, lq = lane & 15;

  const int qT = wave & 127;
  const int h  = (wave >> 7) & 7;
  const int n  = wave >> 10;
  if (n >= 2) return;

  const _Float16* Qb  = Q + (((size_t)n * NH + h) * S_LEN) * HD;
  const _Float16* Kb  = K + (((size_t)n * NH + h) * S_LEN) * HD;
  const _Float16* Vtb = V + (((size_t)n * NH + h) * HD) * S_LEN;

  _Float16* kbuf[2] = {lds + wid * SLAB, lds + wid * SLAB + 2048};
  _Float16* vbuf[2] = {lds + wid * SLAB + 4096, lds + wid * SLAB + 6144};
  _Float16* pl      = lds + wid * SLAB + 8192;

  // Q tile (16 x 64) as two A fragments
  v16h qa0, qa1;
  {
    const _Float16* qrow = Qb + (size_t)(qT * 16 + lq) * HD;
#pragma unroll
    for (int i = 0; i < 16; ++i) {
      const int d = a_col(i, half);
      qa0[i] = qrow[d];
      qa1[i] = qrow[32 + d];
    }
  }

  v8f o0 = {}, o1 = {}, o2 = {}, o3 = {};
  float m[8], sden[8];
#pragma unroll
  for (int i = 0; i < 8; ++i) { m[i] = -1e30f; sden[i] = 0.f; }

  // prologue: DMA chunk 0 into buffer 0
  tdm_load_2d_f16((unsigned)(uintptr_t)kbuf[0], Kb, HD, 32, HD, S_LEN, HD);
  tdm_load_2d_f16((unsigned)(uintptr_t)vbuf[0], Vtb, 32, HD, S_LEN, HD, S_LEN);

  for (int kb = 0; kb < S_LEN; kb += 32) {
    const int cur = (kb >> 5) & 1;
    if (kb + 32 < S_LEN) {
      asm volatile("s_wait_dscnt 0x0" ::: "memory");
      tdm_load_2d_f16((unsigned)(uintptr_t)kbuf[1 ^ cur],
                      Kb + (size_t)(kb + 32) * HD, HD, 32, HD,
                      (unsigned)(S_LEN - kb - 32), HD);
      tdm_load_2d_f16((unsigned)(uintptr_t)vbuf[1 ^ cur],
                      Vtb + (size_t)(kb + 32), 32, HD,
                      (unsigned)(S_LEN - kb - 32), HD, S_LEN);
      __builtin_amdgcn_s_wait_tensorcnt((short)2);  // current chunk landed
    } else {
      __builtin_amdgcn_s_wait_tensorcnt((short)0);
    }
    const _Float16* kl = kbuf[cur];  // [key][d]  32x64
    const _Float16* vl = vbuf[cur];  // [d][key]  64x32

    v8f c0 = {}, c1 = {};
#pragma unroll
    for (int dk = 0; dk < HD; dk += 32) {
      v16h b0, b1;
#pragma unroll
      for (int i = 0; i < 16; ++i) {
        const int d = dk + b_row(i, half);
        b0[i] = kl[lq * HD + d];
        b1[i] = kl[(16 + lq) * HD + d];
      }
      const v16h qa = dk ? qa1 : qa0;
      c0 = wmma_f16(qa, b0, c0);
      c1 = wmma_f16(qa, b1, c1);
    }
    const float sc = 0.125f;  // 1/sqrt(64)
    float p0[8], p1[8], cm[8];
#pragma unroll
    for (int i = 0; i < 8; ++i) {
      p0[i] = (float)c0[i] * sc;
      p1[i] = (float)c1[i] * sc;
      cm[i] = fmaxf(p0[i], p1[i]);
    }
#pragma unroll
    for (int mask = 1; mask < 16; mask <<= 1)
#pragma unroll
      for (int i = 0; i < 8; ++i)
        cm[i] = fmaxf(cm[i], __shfl_xor(cm[i], mask, 32));
    float r[8], ps[8];
#pragma unroll
    for (int i = 0; i < 8; ++i) {
      const float mn = fmaxf(m[i], cm[i]);
      r[i] = __expf(m[i] - mn);
      m[i] = mn;
      p0[i] = __expf(p0[i] - mn);
      p1[i] = __expf(p1[i] - mn);
      ps[i] = p0[i] + p1[i];
    }
#pragma unroll
    for (int mask = 1; mask < 16; mask <<= 1)
#pragma unroll
      for (int i = 0; i < 8; ++i)
        ps[i] += __shfl_xor(ps[i], mask, 32);
#pragma unroll
    for (int i = 0; i < 8; ++i) {
      sden[i] = sden[i] * r[i] + ps[i];
      o0[i] *= r[i]; o1[i] *= r[i]; o2[i] *= r[i]; o3[i] *= r[i];
      pl[(half * 8 + i) * 32 + lq]      = (_Float16)p0[i];
      pl[(half * 8 + i) * 32 + 16 + lq] = (_Float16)p1[i];
    }
    v16h pa;
#pragma unroll
    for (int i = 0; i < 16; ++i) pa[i] = pl[lq * 32 + a_col(i, half)];
#pragma unroll
    for (int j = 0; j < 4; ++j) {
      v16h vb;
#pragma unroll
      for (int i = 0; i < 16; ++i)
        vb[i] = vl[(j * 16 + lq) * 32 + b_row(i, half)];
      v8f& oj = (j == 0) ? o0 : ((j == 1) ? o1 : ((j == 2) ? o2 : o3));
      oj = wmma_f16(pa, vb, oj);
    }
  }
#pragma unroll
  for (int i = 0; i < 8; ++i) {
    const float inv = 1.0f / sden[i];
    const int q = qT * 16 + half * 8 + i;
    _Float16* orow = O + ((size_t)n * S_LEN + q) * TD + h * HD;
    orow[0 * 16 + lq] = (_Float16)((float)o0[i] * inv);
    orow[1 * 16 + lq] = (_Float16)((float)o1[i] * inv);
    orow[2 * 16 + lq] = (_Float16)((float)o2[i] * inv);
    orow[3 * 16 + lq] = (_Float16)((float)o3[i] * inv);
  }
}

// ---------------------------------------------------------------------------
// Kernel 3: FFN layer 1.  hidden = relu(attn_out @ W1.T + b1), f16 weights.
// ---------------------------------------------------------------------------
__global__ void __launch_bounds__(256) ff1_kernel(
    const _Float16* __restrict__ A,    // [4096, 512] f16
    const _Float16* __restrict__ W1h,  // [1024, 512] f16
    const float* __restrict__ b1,
    _Float16* __restrict__ Hd)         // [4096, 1024] f16
{
  const int lane = threadIdx.x & 31;
  const int wave = (int)((blockIdx.x * blockDim.x + threadIdx.x) >> 5);
  const int half = lane >> 4, lq = lane & 15;
  const int rowT = wave & 255;
  const int colT = wave >> 8;
  if (colT >= 64) return;

  v8f acc = {};
  for (int kk = 0; kk < TD; kk += 32) {
    if (kk + 32 < TD)
      __builtin_prefetch(&W1h[(size_t)(colT * 16 + lq) * TD + kk + 32], 0, 3);
    v16h a, b;
#pragma unroll
    for (int i = 0; i < 16; ++i)
      a[i] = A[(size_t)(rowT * 16 + lq) * TD + kk + a_col(i, half)];
#pragma unroll
    for (int i = 0; i < 16; ++i)
      b[i] = W1h[(size_t)(colT * 16 + lq) * TD + kk + b_row(i, half)];
    acc = wmma_f16(a, b, acc);
  }
  const float bias = b1[colT * 16 + lq];
#pragma unroll
  for (int i = 0; i < 8; ++i) {
    float v = (float)acc[i] + bias;
    v = v > 0.f ? v : 0.f;  // ReLU
    Hd[(size_t)(rowT * 16 + half * 8 + i) * FFD + colT * 16 + lq] = (_Float16)v;
  }
}

// ---------------------------------------------------------------------------
// Kernel 4: FFN layer 2 + residual + transpose back to [N, C, H, W].
// ---------------------------------------------------------------------------
__global__ void __launch_bounds__(256) ff2_kernel(
    const _Float16* __restrict__ Hd,   // [4096, 1024] f16
    const _Float16* __restrict__ W2h,  // [64, 1024] f16
    const float* __restrict__ b2,
    const float* __restrict__ x,       // [2, 64, 2048]
    float* __restrict__ out)           // [2, 64, 2048]
{
  const int lane = threadIdx.x & 31;
  const int wave = (int)((blockIdx.x * blockDim.x + threadIdx.x) >> 5);
  const int half = lane >> 4, lq = lane & 15;
  const int rowT = wave & 255;
  const int colT = wave >> 8;
  if (colT >= 4) return;

  v8f acc = {};
  for (int kk = 0; kk < FFD; kk += 32) {
    v16h a, b;
#pragma unroll
    for (int i = 0; i < 16; ++i)
      a[i] = Hd[(size_t)(rowT * 16 + lq) * FFD + kk + a_col(i, half)];
#pragma unroll
    for (int i = 0; i < 16; ++i)
      b[i] = W2h[(size_t)(colT * 16 + lq) * FFD + kk + b_row(i, half)];
    acc = wmma_f16(a, b, acc);
  }
  const int c = colT * 16 + lq;
  const float bias = b2[c];
#pragma unroll
  for (int i = 0; i < 8; ++i) {
    const int row = rowT * 16 + half * 8 + i;
    const int n = row >> 11, s = row & 2047;
    const size_t idx = ((size_t)n * CH + c) * S_LEN + s;
    out[idx] = (float)acc[i] + bias + x[idx];  // residual
  }
}

extern "C" void kernel_launch(void* const* d_in, const int* in_sizes, int n_in,
                              void* d_out, int out_size, void* d_ws, size_t ws_size,
                              hipStream_t stream) {
  const float* x  = (const float*)d_in[0];
  const float* Wq = (const float*)d_in[1];
  const float* bq = (const float*)d_in[2];
  const float* Wk = (const float*)d_in[3];
  const float* bk = (const float*)d_in[4];
  const float* Wv = (const float*)d_in[5];
  const float* bv = (const float*)d_in[6];
  const float* W1 = (const float*)d_in[7];
  const float* b1 = (const float*)d_in[8];
  const float* W2 = (const float*)d_in[9];
  const float* b2 = (const float*)d_in[10];
  float* out = (float*)d_out;

  _Float16* ws = (_Float16*)d_ws;
  const size_t QKV_E = (size_t)2 * NH * S_LEN * HD;  // 2,097,152
  _Float16* Qh  = ws;
  _Float16* Kh  = Qh + QKV_E;
  _Float16* Vh  = Kh + QKV_E;                   // transposed [n,h,d,s]
  _Float16* Oh  = Vh + QKV_E;                   // [2, 2048, 512]
  _Float16* Hh  = Oh + (size_t)2 * S_LEN * TD;  // [4096, 1024]
  _Float16* Wqh = Hh + (size_t)4096 * FFD;      // f16 weights
  _Float16* Wkh = Wqh + (size_t)TD * CH;
  _Float16* Wvh = Wkh + (size_t)TD * CH;
  _Float16* W1h = Wvh + (size_t)TD * CH;
  _Float16* W2h = W1h + (size_t)FFD * TD;

  // 0) one-shot weight conversion (x4 vectorized; sizes are multiples of 1024)
  cvt_kernel<<<32, 256, 0, stream>>>(Wq, Wqh, TD * CH);
  cvt_kernel<<<32, 256, 0, stream>>>(Wk, Wkh, TD * CH);
  cvt_kernel<<<32, 256, 0, stream>>>(Wv, Wvh, TD * CH);
  cvt_kernel<<<512, 256, 0, stream>>>(W1, W1h, FFD * TD);
  cvt_kernel<<<64, 256, 0, stream>>>(W2, W2h, CH * FFD);

  // 1) QKV: one block per (n, which, row tile) = 2*3*128 = 768 blocks
  qkv_kernel<<<768, 256, 0, stream>>>(x, Wqh, bq, Wkh, bk, Wvh, bv, Qh, Kh, Vh);
  // 2) flash attention: 2*8*128 = 2048 waves
  attn_kernel<<<1024, AW * 32, 0, stream>>>(Qh, Kh, Vh, Oh);
  // 3) FFN1: 256*64 = 16384 waves
  ff1_kernel<<<2048, 256, 0, stream>>>(Oh, W1h, b1, Hh);
  // 4) FFN2 + residual: 256*4 = 1024 waves
  ff2_kernel<<<128, 256, 0, stream>>>(Hh, W2h, b2, x, out);
}
